// DecoderLayer_60060822667266
// MI455X (gfx1250) — matrix operand order repaired and checked by reading, not compiled, not run on hardware
//
#include <hip/hip_runtime.h>
#include <hip/hip_bf16.h>
#include <math.h>

// ---------------- problem constants (from reference) ----------------
#define B_SZ   64
#define T_SZ   64
#define E_SZ   2048
#define G_SZ   4096
#define NHEAD  32
#define CHEAD  64
#define SEQ    256          // IMG_TOKENS
#define QSCALE 0.125f       // 1/sqrt(64)

typedef __attribute__((ext_vector_type(16))) __bf16 bf16x16;
typedef __attribute__((ext_vector_type(2)))  __bf16 bf16x2;
typedef __attribute__((ext_vector_type(2)))  float  f32x2;
typedef __attribute__((ext_vector_type(8)))  float  f32x8;

// packed fp32x2 -> bf16x2 (single v_cvt_pk_bf16_f32)
__device__ __forceinline__ unsigned pack2bf(float lo, float hi) {
  f32x2 f; f.x = lo; f.y = hi;
  union { bf16x2 h; unsigned u; } p;
  p.h = __builtin_convertvector(f, bf16x2);
  return p.u;
}

// =====================================================================
// GEMM  C[M,N] (+partial) = A[M,K] @ W[K,N], fp32 in/out, bf16 WMMA core.
// BM=BN=64, BK=64. 256 threads = 8 wave32; wave owns one 16-row stripe
// and two 16x16 C tiles -> 4 v_wmma_f32_16x16x32_bf16 per K tile.
// Double-buffered LDS (32KB) + register prefetch: global b128 loads for
// tile t+1 are in flight while tile t's WMMAs run; ONE barrier per tile.
// Split-K: blockIdx.z processes kLen of K and writes partial z*M*N.
// =====================================================================
#define BM 64
#define BN 64
#define BK 64

__global__ __launch_bounds__(256) void gemm_bf16_wmma(
    const float* __restrict__ A, const float* __restrict__ W,
    float* __restrict__ C, int M, int N, int K, int kLen)
{
  __shared__ __align__(16) __bf16 lA[2][BM][BK];   // row-major (m,k)   16KB
  __shared__ __align__(16) __bf16 lB[2][BN][BK];   // transposed (n,k)  16KB

  const int tid  = threadIdx.x;
  const int lane = tid & 31;
  const int wave = tid >> 5;
  const int m0 = blockIdx.y * BM;
  const int n0 = blockIdx.x * BN;
  const int kbase = blockIdx.z * kLen;
  C += (size_t)blockIdx.z * M * N;                 // split-K partial slab

  const int wm = (wave & 3) * 16;     // wave's M tile inside block
  const int wn = (wave >> 2) * 32;    // wave's first N tile inside block
  const int sel = lane >> 4;          // lane half (ISA 16-bit frag layout)
  const int r   = lane & 15;

  // staging decomposition
  const int a_row = tid >> 2;         // 0..63
  const int a_kc  = (tid & 3) * 16;   // 0,16,32,48
  const int w_k   = (tid >> 4) * 4;   // 0,4,...,60   (4 consecutive k rows)
  const int w_n   = (tid & 15) * 4;   // 0,4,...,60   (4 consecutive n cols)

  f32x8 c0 = {}; f32x8 c1 = {};
  float pa[16], pw[16];               // register prefetch staging

  const int NK = kLen >> 6;

  // ---- prefetch K-tile 0 into registers ----
  {
    const float* ap = A + (size_t)(m0 + a_row) * K + kbase + a_kc;
    const float* wp = W + (size_t)(kbase + w_k) * N + n0 + w_n;
    #pragma unroll
    for (int j = 0; j < 4; ++j)
      *reinterpret_cast<float4*>(&pa[4 * j]) = *reinterpret_cast<const float4*>(ap + 4 * j);
    #pragma unroll
    for (int j = 0; j < 4; ++j)
      *reinterpret_cast<float4*>(&pw[4 * j]) = *reinterpret_cast<const float4*>(wp + (size_t)j * N);
  }

  int buf = 0;
  for (int kt = 0; kt < NK; ++kt) {
    // ---- dump prefetched registers into LDS[buf] (b128 / b64 stores) ----
    {
      union { unsigned u[8]; uint4 q[2]; } ab;
      #pragma unroll
      for (int j = 0; j < 8; ++j) ab.u[j] = pack2bf(pa[2 * j], pa[2 * j + 1]);
      *reinterpret_cast<uint4*>(&lA[buf][a_row][a_kc])     = ab.q[0];
      *reinterpret_cast<uint4*>(&lA[buf][a_row][a_kc + 8]) = ab.q[1];
      #pragma unroll
      for (int c = 0; c < 4; ++c) {            // transpose: lB[n][k]
        uint2 d;
        d.x = pack2bf(pw[c],     pw[4 + c]);   // k, k+1
        d.y = pack2bf(pw[8 + c], pw[12 + c]);  // k+2, k+3
        *reinterpret_cast<uint2*>(&lB[buf][w_n + c][w_k]) = d;
      }
    }
    __syncthreads();   // single barrier per K-tile

    // ---- issue next tile's global loads; they fly during the WMMAs ----
    if (kt + 1 < NK) {
      const int kk = kbase + ((kt + 1) << 6);
      const float* ap = A + (size_t)(m0 + a_row) * K + kk + a_kc;
      const float* wp = W + (size_t)(kk + w_k) * N + n0 + w_n;
      #pragma unroll
      for (int j = 0; j < 4; ++j)
        *reinterpret_cast<float4*>(&pa[4 * j]) = *reinterpret_cast<const float4*>(ap + 4 * j);
      #pragma unroll
      for (int j = 0; j < 4; ++j)
        *reinterpret_cast<float4*>(&pw[4 * j]) = *reinterpret_cast<const float4*>(wp + (size_t)j * N);
    }

    // ---- 4 WMMAs from LDS[buf] (ISA 7.12.2 16-bit fragment layouts) ----
    #pragma unroll
    for (int ks = 0; ks < BK; ks += 32) {
      union Frag { bf16x16 v; uint4 q[2]; } fa, fb0, fb1;
      // A 16x32: lanes 0-15 K=0..15, lanes 16-31 K=16..31 (contiguous, matches B)
      fa.q[0]  = *reinterpret_cast<const uint4*>(&lA[buf][wm + r][ks + sel * 16]);
      fa.q[1]  = *reinterpret_cast<const uint4*>(&lA[buf][wm + r][ks + sel * 16 + 8]);
      // B 32x16: lane half selects K 0..15 / 16..31, contiguous per column
      fb0.q[0] = *reinterpret_cast<const uint4*>(&lB[buf][wn + r][ks + sel * 16]);
      fb0.q[1] = *reinterpret_cast<const uint4*>(&lB[buf][wn + r][ks + sel * 16 + 8]);
      fb1.q[0] = *reinterpret_cast<const uint4*>(&lB[buf][wn + 16 + r][ks + sel * 16]);
      fb1.q[1] = *reinterpret_cast<const uint4*>(&lB[buf][wn + 16 + r][ks + sel * 16 + 8]);

      c0 = __builtin_amdgcn_wmma_f32_16x16x32_bf16(false, fa.v, false, fb0.v,
                                                   (short)0, c0, false, false);
      c1 = __builtin_amdgcn_wmma_f32_16x16x32_bf16(false, fa.v, false, fb1.v,
                                                   (short)0, c1, false, false);
    }
    buf ^= 1;
  }

  // ---- store C (f32 16x16 layout: vgpr i -> M = i + sel*8, N = r) ----
  const int cm = m0 + wm + sel * 8;
  #pragma unroll
  for (int i = 0; i < 8; ++i) {
    C[(size_t)(cm + i) * N + n0 + wn + r]      = c0[i];
    C[(size_t)(cm + i) * N + n0 + wn + 16 + r] = c1[i];
  }
}

// fixed-order split-K reduction (bit-deterministic across replays)
__global__ __launch_bounds__(256) void reduce_k_kernel(
    const float* __restrict__ part, float* __restrict__ out, size_t mn, int Z)
{
  size_t i = (size_t)blockIdx.x * 256 + threadIdx.x;
  if (i < mn) {
    float s = 0.f;
    for (int z = 0; z < Z; ++z) s += part[i + (size_t)z * mn];
    out[i] = s;
  }
}

// =====================================================================
// LayerNorm over last dim (population variance, matches jnp.var)
// =====================================================================
__global__ __launch_bounds__(256) void ln_kernel(
    const float* __restrict__ x, const float* __restrict__ g,
    const float* __restrict__ b, float* __restrict__ y, int cols)
{
  const int row = blockIdx.x;
  const float* xr = x + (size_t)row * cols;
  float* yr = y + (size_t)row * cols;
  float s = 0.f, ss = 0.f;
  for (int i = threadIdx.x; i < cols; i += 256) { float v = xr[i]; s += v; ss += v * v; }
  __shared__ float rs[256], rss[256];
  rs[threadIdx.x] = s; rss[threadIdx.x] = ss;
  __syncthreads();
  for (int o = 128; o > 0; o >>= 1) {
    if (threadIdx.x < o) { rs[threadIdx.x] += rs[threadIdx.x + o]; rss[threadIdx.x] += rss[threadIdx.x + o]; }
    __syncthreads();
  }
  const float mean = rs[0] / (float)cols;
  const float var  = rss[0] / (float)cols - mean * mean;
  const float inv  = rsqrtf(var + 1e-5f);
  for (int i = threadIdx.x; i < cols; i += 256)
    yr[i] = (xr[i] - mean) * inv * g[i] + b[i];
}

// =====================================================================
// elementwise helpers
// =====================================================================
__global__ void add_kernel(const float* __restrict__ a, const float* __restrict__ b,
                           float* __restrict__ o, int n)
{
  int i = blockIdx.x * blockDim.x + threadIdx.x;
  if (i < n) o[i] = a[i] + b[i];
}

__global__ void gelu_mul_kernel(const float* __restrict__ a, const float* __restrict__ b,
                                float* __restrict__ o, int n)
{
  int i = blockIdx.x * blockDim.x + threadIdx.x;
  if (i < n) {
    float x = a[i];
    float g = 0.5f * x * (1.0f + erff(x * 0.7071067811865475f));  // exact gelu
    o[i] = g * b[i];
  }
}

// 536 MB cache passthrough: float4 streaming copy (memory-roofline part)
__global__ __launch_bounds__(256) void copy_cache_kernel(
    const float4* __restrict__ src, float4* __restrict__ dst, size_t n4)
{
  size_t i = (size_t)blockIdx.x * blockDim.x + threadIdx.x;
  if (i < n4) dst[i] = src[i];
}

// write new K/V rows at token_index into the output cache
__global__ void scatter_kv_kernel(const float* __restrict__ kN, const float* __restrict__ vN,
                                  float* __restrict__ cacheOut, const int* __restrict__ tokp)
{
  const int b = blockIdx.x;
  const int tok = *tokp;
  float* kd = cacheOut + ((size_t)b * SEQ + tok) * E_SZ;
  float* vd = cacheOut + ((size_t)(B_SZ + b) * SEQ + tok) * E_SZ;
  for (int i = threadIdx.x; i < E_SZ; i += blockDim.x) {
    kd[i] = kN[(size_t)b * E_SZ + i];
    vd[i] = vN[(size_t)b * E_SZ + i];
  }
}

// =====================================================================
// Self-attention, one (batch, head) per block. Sq=1, Sk=256 (causal to
// token_index). K/V rows < tok come from the *input* cache; row == tok
// comes from the fresh projections (cache update semantics).
// =====================================================================
__global__ __launch_bounds__(128) void self_attn_kernel(
    const float* __restrict__ q, const float* __restrict__ kN,
    const float* __restrict__ vN, const float* __restrict__ cache,
    const int* __restrict__ tokp, float* __restrict__ out)
{
  const int b = blockIdx.x, h = blockIdx.y;
  const int tok = *tokp;
  const int tid = threadIdx.x;
  __shared__ float qs[CHEAD];
  __shared__ float w[SEQ];
  __shared__ float red[128];

  if (tid < CHEAD) qs[tid] = q[(size_t)b * E_SZ + h * CHEAD + tid] * QSCALE;
  __syncthreads();

  for (int k = tid; k < SEQ; k += 128) {
    float sc = -1e12f;                         // (1-mask)*-1e12 bias
    if (k <= tok) {
      const float* kr = (k == tok) ? (kN + (size_t)b * E_SZ + h * CHEAD)
                                   : (cache + ((size_t)b * SEQ + k) * E_SZ + h * CHEAD);
      float d = 0.f;
      #pragma unroll 8
      for (int c = 0; c < CHEAD; ++c) d += qs[c] * kr[c];
      sc = d;
    }
    w[k] = sc;
  }
  __syncthreads();

  float m = -1e30f;
  for (int k = tid; k < SEQ; k += 128) m = fmaxf(m, w[k]);
  red[tid] = m; __syncthreads();
  for (int o = 64; o > 0; o >>= 1) { if (tid < o) red[tid] = fmaxf(red[tid], red[tid + o]); __syncthreads(); }
  m = red[0]; __syncthreads();

  float s = 0.f;
  for (int k = tid; k < SEQ; k += 128) { float e = expf(w[k] - m); w[k] = e; s += e; }
  red[tid] = s; __syncthreads();
  for (int o = 64; o > 0; o >>= 1) { if (tid < o) red[tid] += red[tid + o]; __syncthreads(); }
  const float inv = 1.f / red[0];
  __syncthreads();

  for (int c = tid; c < CHEAD; c += 128) {
    float acc = 0.f;
    for (int k = 0; k <= tok; ++k) {
      const float* vr = (k == tok) ? (vN + (size_t)b * E_SZ + h * CHEAD)
                                   : (cache + ((size_t)(B_SZ + b) * SEQ + k) * E_SZ + h * CHEAD);
      acc += w[k] * vr[c];
    }
    out[(size_t)b * E_SZ + h * CHEAD + c] = acc * inv;
  }
}

// =====================================================================
// Cross-attention, one (batch, head) per block. Sq=1, Sk=T=64.
// =====================================================================
__global__ __launch_bounds__(128) void cross_attn_kernel(
    const float* __restrict__ q, const float* __restrict__ K,
    const float* __restrict__ V, const unsigned char* __restrict__ mask,
    float* __restrict__ out)
{
  const int b = blockIdx.x, h = blockIdx.y;
  const int tid = threadIdx.x;
  __shared__ float qs[CHEAD];
  __shared__ float w[T_SZ];
  __shared__ float red[128];

  if (tid < CHEAD) qs[tid] = q[(size_t)b * E_SZ + h * CHEAD + tid] * QSCALE;
  __syncthreads();

  if (tid < T_SZ) {
    const float* kr = K + ((size_t)b * T_SZ + tid) * E_SZ + h * CHEAD;
    float d = 0.f;
    #pragma unroll 8
    for (int c = 0; c < CHEAD; ++c) d += qs[c] * kr[c];
    float bias = mask[(size_t)b * T_SZ + tid] ? 0.f : -1e12f;
    w[tid] = d + bias;
  }
  __syncthreads();

  float m = (tid < T_SZ) ? w[tid] : -1e30f;
  red[tid] = m; __syncthreads();
  for (int o = 64; o > 0; o >>= 1) { if (tid < o) red[tid] = fmaxf(red[tid], red[tid + o]); __syncthreads(); }
  m = red[0]; __syncthreads();

  float s = 0.f;
  if (tid < T_SZ) { s = expf(w[tid] - m); w[tid] = s; }
  red[tid] = s; __syncthreads();
  for (int o = 64; o > 0; o >>= 1) { if (tid < o) red[tid] += red[tid + o]; __syncthreads(); }
  const float inv = 1.f / red[0];
  __syncthreads();

  for (int c = tid; c < CHEAD; c += 128) {
    float acc = 0.f;
    for (int k = 0; k < T_SZ; ++k)
      acc += w[k] * V[((size_t)b * T_SZ + k) * E_SZ + h * CHEAD + c];
    out[(size_t)b * E_SZ + h * CHEAD + c] = acc * inv;
  }
}

// =====================================================================
// driver
// =====================================================================
#define KSPLIT 4

static inline void launch_gemm(const float* A, const float* W, float* C,
                               int M, int N, int K, int Z, float* part,
                               hipStream_t stream)
{
  dim3 g(N / BN, M / BM, Z);
  gemm_bf16_wmma<<<g, dim3(256), 0, stream>>>(A, W, (Z == 1) ? C : part,
                                              M, N, K, K / Z);
  if (Z > 1) {
    size_t mn = (size_t)M * N;
    reduce_k_kernel<<<dim3((unsigned)((mn + 255) / 256)), dim3(256), 0, stream>>>(
        part, C, mn, Z);
  }
}

extern "C" void kernel_launch(void* const* d_in, const int* in_sizes, int n_in,
                              void* d_out, int out_size, void* d_ws, size_t ws_size,
                              hipStream_t stream)
{
  const float* dec   = (const float*)d_in[0];             // (64,1,2048)
  const float* enc   = (const float*)d_in[1];             // (64,64,2048)
  const float* cache = (const float*)d_in[2];             // (128,256,2048)
  const unsigned char* amask = (const unsigned char*)d_in[3]; // (64,64) bool
  const int*   tok   = (const int*)d_in[4];
  const float* ln_pre_sa_g = (const float*)d_in[5];
  const float* ln_pre_sa_b = (const float*)d_in[6];
  const float* ln_sa_g     = (const float*)d_in[7];
  const float* ln_sa_b     = (const float*)d_in[8];
  const float* ln_pre_ca_g = (const float*)d_in[9];
  const float* ln_pre_ca_b = (const float*)d_in[10];
  const float* ln_ca_g     = (const float*)d_in[11];
  const float* ln_ca_b     = (const float*)d_in[12];
  const float* sa_wq = (const float*)d_in[13];
  const float* sa_wk = (const float*)d_in[14];
  const float* sa_wv = (const float*)d_in[15];
  const float* sa_wo = (const float*)d_in[16];
  const float* ca_wq = (const float*)d_in[17];
  const float* ca_wk = (const float*)d_in[18];
  const float* ca_wv = (const float*)d_in[19];
  const float* ca_wo = (const float*)d_in[20];
  const float* glu_ln0_g = (const float*)d_in[21];
  const float* glu_ln0_b = (const float*)d_in[22];
  const float* glu_fc0   = (const float*)d_in[23];
  const float* glu_fc1   = (const float*)d_in[24];
  const float* glu_ln1_g = (const float*)d_in[25];
  const float* glu_ln1_b = (const float*)d_in[26];
  const float* glu_fc2   = (const float*)d_in[27];

  float* out_x     = (float*)d_out;                        // (64,1,2048)
  float* out_cache = (float*)d_out + (size_t)B_SZ * E_SZ;  // (128,256,2048)

  // ---- workspace layout (floats) ----
  float* w = (float*)d_ws;
  const size_t BE = (size_t)B_SZ * E_SZ;        // 131072
  const size_t BG = (size_t)B_SZ * G_SZ;        // 262144
  const size_t TE = (size_t)B_SZ * T_SZ * E_SZ; // 8388608
  float* xln   = w;            w += BE;
  float* kN    = w;            w += BE;
  float* vN    = w;            w += BE;
  float* qN    = w;            w += BE;
  float* attn  = w;            w += BE;
  float* proj  = w;            w += BE;
  float* projl = w;            w += BE;
  float* x1    = w;            w += BE;
  float* yln   = w;            w += BE;
  float* qC    = w;            w += BE;
  float* cattn = w;            w += BE;
  float* x2    = w;            w += BE;
  float* zln   = w;            w += BE;
  float* z2    = w;            w += BE;
  float* a0    = w;            w += BG;
  float* a1    = w;            w += BG;
  float* gm    = w;            w += BG;
  float* gmln  = w;            w += BG;
  float* gpart = w;            w += (size_t)KSPLIT * BG;   // split-K partials
  float* Kc    = w;            w += TE;
  float* Vc    = w;            w += TE;
  (void)ws_size; (void)in_sizes; (void)n_in; (void)out_size;

  const dim3 blk256(256), blk128(128);
  const dim3 gBH(B_SZ, NHEAD);

  // 1) stream the KV cache through to the output (bulk of the bytes)
  {
    size_t n4 = (size_t)2 * B_SZ * SEQ * E_SZ / 4;
    copy_cache_kernel<<<dim3((unsigned)(n4 / 256)), blk256, 0, stream>>>(
        (const float4*)cache, (float4*)out_cache, n4);
  }

  // 2) self-attention branch
  ln_kernel<<<dim3(B_SZ), blk256, 0, stream>>>(dec, ln_pre_sa_g, ln_pre_sa_b, xln, E_SZ);
  launch_gemm(xln, sa_wk, kN, B_SZ, E_SZ, E_SZ, KSPLIT, gpart, stream);
  launch_gemm(xln, sa_wv, vN, B_SZ, E_SZ, E_SZ, KSPLIT, gpart, stream);
  launch_gemm(xln, sa_wq, qN, B_SZ, E_SZ, E_SZ, KSPLIT, gpart, stream);
  scatter_kv_kernel<<<dim3(B_SZ), blk256, 0, stream>>>(kN, vN, out_cache, tok);
  self_attn_kernel<<<gBH, blk128, 0, stream>>>(qN, kN, vN, cache, tok, attn);
  launch_gemm(attn, sa_wo, proj, B_SZ, E_SZ, E_SZ, KSPLIT, gpart, stream);
  ln_kernel<<<dim3(B_SZ), blk256, 0, stream>>>(proj, ln_sa_g, ln_sa_b, projl, E_SZ);
  add_kernel<<<dim3((unsigned)(BE / 256)), blk256, 0, stream>>>(dec, projl, x1, (int)BE);

  // 3) cross-attention branch
  ln_kernel<<<dim3(B_SZ), blk256, 0, stream>>>(x1, ln_pre_ca_g, ln_pre_ca_b, yln, E_SZ);
  launch_gemm(yln, ca_wq, qC, B_SZ, E_SZ, E_SZ, KSPLIT, gpart, stream);
  launch_gemm(enc, ca_wk, Kc, B_SZ * T_SZ, E_SZ, E_SZ, 1, nullptr, stream);
  launch_gemm(enc, ca_wv, Vc, B_SZ * T_SZ, E_SZ, E_SZ, 1, nullptr, stream);
  cross_attn_kernel<<<gBH, blk128, 0, stream>>>(qC, Kc, Vc, amask, cattn);
  launch_gemm(cattn, ca_wo, proj, B_SZ, E_SZ, E_SZ, KSPLIT, gpart, stream);
  ln_kernel<<<dim3(B_SZ), blk256, 0, stream>>>(proj, ln_ca_g, ln_ca_b, projl, E_SZ);
  add_kernel<<<dim3((unsigned)(BE / 256)), blk256, 0, stream>>>(x1, projl, x2, (int)BE);

  // 4) GLU MLP
  ln_kernel<<<dim3(B_SZ), blk256, 0, stream>>>(x2, glu_ln0_g, glu_ln0_b, zln, E_SZ);
  launch_gemm(zln, glu_fc0, a0, B_SZ, G_SZ, E_SZ, KSPLIT, gpart, stream);
  launch_gemm(zln, glu_fc1, a1, B_SZ, G_SZ, E_SZ, KSPLIT, gpart, stream);
  gelu_mul_kernel<<<dim3((unsigned)(BG / 256)), blk256, 0, stream>>>(a0, a1, gm, (int)BG);
  ln_kernel<<<dim3(B_SZ), blk256, 0, stream>>>(gm, glu_ln1_g, glu_ln1_b, gmln, G_SZ);
  launch_gemm(gmln, glu_fc2, z2, B_SZ, E_SZ, G_SZ, KSPLIT, gpart, stream);
  add_kernel<<<dim3((unsigned)(BE / 256)), blk256, 0, stream>>>(x2, z2, out_x, (int)BE);
}
